// ami_only_net_82832739270724
// MI455X (gfx1250) — matrix operand order repaired and checked by reading, not compiled, not run on hardware
//
#include <hip/hip_runtime.h>
#include <hip/hip_bf16.h>

// ---------------------------------------------------------------------------
// Problem constants (match reference setup_inputs)
// ---------------------------------------------------------------------------
#define N1 50000
#define E1 200000
#define N3 20000
#define E3 160000
#define BGR 256         // graphs per batch
#define F_XD 52
#define D1 128
#define D2 64
#define D3 1900
#define D4 1024
#define D5 256
#define NEG_SLOPE 0.2f

#define CDIV(a, b) (((a) + (b) - 1) / (b))

typedef __attribute__((ext_vector_type(16))) __bf16 bf16x16;
typedef __attribute__((ext_vector_type(8)))  __bf16 bf16x8;
typedef __attribute__((ext_vector_type(8)))  float  f32x8;

__device__ __forceinline__ bf16x8 pack8(float4 x, float4 y) {
  bf16x8 r;
  r[0] = (__bf16)x.x; r[1] = (__bf16)x.y; r[2] = (__bf16)x.z; r[3] = (__bf16)x.w;
  r[4] = (__bf16)y.x; r[5] = (__bf16)y.y; r[6] = (__bf16)y.z; r[7] = (__bf16)y.w;
  return r;
}

// Build a 16-element fragment from two 16B LDS chunks of one tile row.
__device__ __forceinline__ bf16x16 frag_from(const __bf16 (*T)[40], int r,
                                             int off0, int off1) {
  bf16x8 x0 = *reinterpret_cast<const bf16x8*>(&T[r][off0]);
  bf16x8 x1 = *reinterpret_cast<const bf16x8*>(&T[r][off1]);
  bf16x16 f;
#pragma unroll
  for (int i = 0; i < 8; ++i) { f[i] = x0[i]; f[i + 8] = x1[i]; }
  return f;
}

// ---------------------------------------------------------------------------
// Tiled GEMM: C[M,N] = (A[M,K] @ W[K,N]) (+bias) (ReLU optional)
// fp32 in/out, bf16 LDS tiles, v_wmma_f32_16x16x32_bf16 accumulate.
// Block = 128 threads (4 waves). Block tile 64x64; each wave owns a 32x32
// quadrant = 2x2 WMMA accumulators (4 wmma / wave / k-step, fragments reused
// 2x). Register-prefetch software pipeline hides global latency behind WMMAs.
// Edge blocks and K-tails use clamped-address + select loads (branch-free).
// NOTE: call sites guarantee K % 4 == 0 and 16B-aligned A rows.
// ---------------------------------------------------------------------------
__global__ __launch_bounds__(128)
void gemm_bf16_wmma(const float* __restrict__ A, const float* __restrict__ W,
                    const float* __restrict__ bias, float* __restrict__ C,
                    int M, int K, int N, int do_relu)
{
  __shared__ alignas(16) __bf16 As[64][40];   // [m][k], 80B row stride
  __shared__ alignas(16) __bf16 Bs[64][40];   // [n][k] (W tile transposed)

  const int t    = threadIdx.x;
  const int lane = t & 31;
  const int wid  = t >> 5;             // 0..3
  const int wy   = wid >> 1;           // wave quadrant row
  const int wx   = wid & 1;            // wave quadrant col
  const int m0   = blockIdx.y * 64;
  const int n0   = blockIdx.x * 64;
  const int hi   = lane >> 4;          // 0/1
  const int l16  = lane & 15;

  // cooperative-load indices
  const int arow = t >> 1;             // A: row in tile (0..63)
  const int akk  = (t & 1) * 16;       // A: k offset (16 consecutive)
  const int bn   = t & 63;             // W: column in tile (coalesced lanes)
  const int bkr  = (t >> 6) * 16;      // W: k row start (16 consecutive rows)

  f32x8 acc00 = {0.f,0.f,0.f,0.f,0.f,0.f,0.f,0.f};
  f32x8 acc01 = acc00, acc10 = acc00, acc11 = acc00;

  const bool interior = (m0 + 64 <= M) && (n0 + 64 <= N);
  const int  Kf       = interior ? (K & ~31) : 0;

  bf16x8 av0, av1, bv0, bv1;

  // ---- fast loader (no guards, vectorized) ----
  auto load_fast = [&](int k0) {
    const float* ap = &A[(size_t)(m0 + arow) * K + k0 + akk];
    float4 a0 = *reinterpret_cast<const float4*>(ap);
    float4 a1 = *reinterpret_cast<const float4*>(ap + 4);
    float4 a2 = *reinterpret_cast<const float4*>(ap + 8);
    float4 a3 = *reinterpret_cast<const float4*>(ap + 12);
    av0 = pack8(a0, a1);
    av1 = pack8(a2, a3);
    const float* wp = &W[(size_t)(k0 + bkr) * N + n0 + bn];
    float wv[16];
#pragma unroll
    for (int i = 0; i < 16; ++i) wv[i] = wp[(size_t)i * N];
#pragma unroll
    for (int i = 0; i < 8; ++i) { bv0[i] = (__bf16)wv[i]; bv1[i] = (__bf16)wv[i + 8]; }
  };

  // ---- guarded loader: clamped addresses + select (branch-free) ----
  auto load_guard = [&](int k0) {
    const int gm  = m0 + arow;
    const int gmc = gm < M - 1 ? gm : M - 1;
#pragma unroll
    for (int i = 0; i < 16; ++i) {
      const int gk  = k0 + akk + i;
      const int gkc = gk < K - 1 ? gk : K - 1;
      float v = A[(size_t)gmc * K + gkc];
      v = (gm < M && gk < K) ? v : 0.0f;
      if (i < 8) av0[i] = (__bf16)v; else av1[i - 8] = (__bf16)v;
    }
    const int gn  = n0 + bn;
    const int gnc = gn < N - 1 ? gn : N - 1;
#pragma unroll
    for (int i = 0; i < 16; ++i) {
      const int gk  = k0 + bkr + i;
      const int gkc = gk < K - 1 ? gk : K - 1;
      float v = W[(size_t)gkc * N + gnc];
      v = (gk < K && gn < N) ? v : 0.0f;
      if (i < 8) bv0[i] = (__bf16)v; else bv1[i - 8] = (__bf16)v;
    }
  };

  auto stage = [&]() {
    *reinterpret_cast<bf16x8*>(&As[arow][akk])     = av0;
    *reinterpret_cast<bf16x8*>(&As[arow][akk + 8]) = av1;
    *reinterpret_cast<bf16x8*>(&Bs[bn][bkr])       = bv0;
    *reinterpret_cast<bf16x8*>(&Bs[bn][bkr + 8])   = bv1;
  };

  auto mma = [&]() {
    const int klo = 16 * hi;            // A frag K base
    const int kb  = 16 * hi;            // B frag K base
    bf16x16 af0 = frag_from(As, 32 * wy + l16,      klo, klo + 8);
    bf16x16 af1 = frag_from(As, 32 * wy + 16 + l16, klo, klo + 8);
    bf16x16 bf0 = frag_from(Bs, 32 * wx + l16,      kb,  kb + 8);
    bf16x16 bf1 = frag_from(Bs, 32 * wx + 16 + l16, kb,  kb + 8);
    acc00 = __builtin_amdgcn_wmma_f32_16x16x32_bf16(false, af0, false, bf0,
                                                    (short)0, acc00, false, false);
    acc01 = __builtin_amdgcn_wmma_f32_16x16x32_bf16(false, af0, false, bf1,
                                                    (short)0, acc01, false, false);
    acc10 = __builtin_amdgcn_wmma_f32_16x16x32_bf16(false, af1, false, bf0,
                                                    (short)0, acc10, false, false);
    acc11 = __builtin_amdgcn_wmma_f32_16x16x32_bf16(false, af1, false, bf1,
                                                    (short)0, acc11, false, false);
  };

  int k0 = 0;
  if (Kf > 0) {
    load_fast(0);                               // prologue
    for (k0 = 0; k0 < Kf; k0 += 32) {
      stage();
      __syncthreads();
      const bool more = (k0 + 32 < Kf);
      bf16x8 sa0 = av0, sa1 = av1, sb0 = bv0, sb1 = bv1; (void)sa0;(void)sa1;(void)sb0;(void)sb1;
      if (more) load_fast(k0 + 32);             // prefetch overlaps WMMAs
      mma();
      __syncthreads();
    }
  }
  for (; k0 < K; k0 += 32) {                    // edge blocks / K tail
    load_guard(k0);
    stage();
    __syncthreads();
    mma();
    __syncthreads();
  }

  // ---- store: quadrant (i,j): row = m0+32*wy+16*i + v + 8*hi,
  //                             col = n0+32*wx+16*j + l16
  const int colBase = n0 + 32 * wx + l16;
  if (interior) {
#pragma unroll
    for (int j = 0; j < 2; ++j) {
      const int col = colBase + 16 * j;
      const float bvv = bias ? bias[col] : 0.0f;
#pragma unroll
      for (int i = 0; i < 2; ++i) {
        const f32x8 a = (i == 0) ? (j == 0 ? acc00 : acc01)
                                 : (j == 0 ? acc10 : acc11);
#pragma unroll
        for (int v = 0; v < 8; ++v) {
          const int row = m0 + 32 * wy + 16 * i + v + 8 * hi;
          float r = a[v] + bvv;
          if (do_relu) r = fmaxf(r, 0.0f);
          C[(size_t)row * N + col] = r;
        }
      }
    }
  } else {
#pragma unroll
    for (int j = 0; j < 2; ++j) {
      const int col = colBase + 16 * j;
      if (col >= N) continue;
      const float bvv = bias ? bias[col] : 0.0f;
#pragma unroll
      for (int i = 0; i < 2; ++i) {
        const f32x8 a = (i == 0) ? (j == 0 ? acc00 : acc01)
                                 : (j == 0 ? acc10 : acc11);
#pragma unroll
        for (int v = 0; v < 8; ++v) {
          const int row = m0 + 32 * wy + 16 * i + v + 8 * hi;
          if (row < M) {
            float r = a[v] + bvv;
            if (do_relu) r = fmaxf(r, 0.0f);
            C[(size_t)row * N + col] = r;
          }
        }
      }
    }
  }
}

// ---------------------------------------------------------------------------
// Utility / graph kernels
// ---------------------------------------------------------------------------
__global__ void zero_f32(float* p, size_t n) {
  size_t i = (size_t)blockIdx.x * 256 + threadIdx.x;
  if (i < n) p[i] = 0.0f;
}

__global__ void relu_copy(const float* in, float* out, size_t n) {
  size_t i = (size_t)blockIdx.x * 256 + threadIdx.x;
  if (i < n) out[i] = fmaxf(in[i], 0.0f);
}

__global__ void deg_edges(const int* __restrict__ dst, const float* __restrict__ ew,
                          float* deg, int E) {
  int e = blockIdx.x * 256 + threadIdx.x;
  if (e >= E) return;
  atomicAdd(&deg[dst[e]], ew ? ew[e] : 1.0f);
}

// deg += 1 (self loop, weight 1), dinv = rsqrt(deg)
__global__ void deg_finalize(const float* deg, float* dinv, int Nn) {
  int i = blockIdx.x * 256 + threadIdx.x;
  if (i >= Nn) return;
  float d = deg[i] + 1.0f;
  dinv[i] = (d > 0.0f) ? rsqrtf(d) : 0.0f;
}

// out[dst,f] += h[src,f] * dinv[src]*ew*dinv[dst]    (edge-parallel, atomics)
__global__ void gcn_edge_agg(const float* __restrict__ h, int hstride,
                             const int* __restrict__ src, const int* __restrict__ dst,
                             const float* __restrict__ ew, const float* __restrict__ dinv,
                             float* out, int ostride, int E, int F) {
  int e = blockIdx.x;
  int f = blockIdx.y * 256 + threadIdx.x;
  if (e >= E || f >= F) return;
  int s = src[e], d = dst[e];
  float w = ew ? ew[e] : 1.0f;
  float norm = dinv[s] * w * dinv[d];
  atomicAdd(&out[(size_t)d * ostride + f], h[(size_t)s * hstride + f] * norm);
}

// out = relu(agg + h*dinv^2 (self loop) + bias)   (in-place on agg)
__global__ void gcn_finalize(const float* __restrict__ h, int hstride,
                             float* out, int ostride,
                             const float* __restrict__ dinv,
                             const float* __restrict__ bias, int Nn, int F) {
  size_t idx = (size_t)blockIdx.x * 256 + threadIdx.x;
  size_t total = (size_t)Nn * F;
  if (idx >= total) return;
  int i = (int)(idx / F);
  int f = (int)(idx % F);
  float di = dinv[i];
  float v = out[(size_t)i * ostride + f] + h[(size_t)i * hstride + f] * di * di + bias[f];
  out[(size_t)i * ostride + f] = fmaxf(v, 0.0f);
}

__global__ void set_col(float* dst, int stride, int col, const float* v, int Nn) {
  int i = blockIdx.x * 256 + threadIdx.x;
  if (i < Nn) dst[(size_t)i * stride + col] = v[i];
}

// ---- GAT helpers -----------------------------------------------------------
__device__ __forceinline__ float leaky(float x) { return x > 0.0f ? x : NEG_SLOPE * x; }
// order-preserving float -> uint encoding for atomicMax
__device__ __forceinline__ unsigned fenc(float x) {
  unsigned u = __float_as_uint(x);
  return (u & 0x80000000u) ? ~u : (u | 0x80000000u);
}
__device__ __forceinline__ float fdec(unsigned u) {
  return __uint_as_float((u & 0x80000000u) ? (u & 0x7fffffffu) : ~u);
}

// per node: a_s = h.att_src, a_d = h.att_dst, init m with self-loop score
__global__ __launch_bounds__(256)
void gat_scores(const float* __restrict__ h, const float* __restrict__ atts,
                const float* __restrict__ attd, float* a_s, float* a_d,
                unsigned* menc, int Nn) {
  __shared__ float red[256];
  int i = blockIdx.x, t = threadIdx.x;
  float hv = h[(size_t)i * 256 + t];
  red[t] = hv * atts[t];
  __syncthreads();
  for (int s = 128; s > 0; s >>= 1) { if (t < s) red[t] += red[t + s]; __syncthreads(); }
  float asv = red[0];
  __syncthreads();
  red[t] = hv * attd[t];
  __syncthreads();
  for (int s = 128; s > 0; s >>= 1) { if (t < s) red[t] += red[t + s]; __syncthreads(); }
  float adv = red[0];
  if (t == 0) {
    a_s[i] = asv;
    a_d[i] = adv;
    menc[i] = fenc(leaky(asv + adv));   // self-loop initializes segment max
  }
}

__global__ void gat_edge_max(const float* a_s, const float* a_d,
                             const int* src, const int* dst, unsigned* menc, int E) {
  int e = blockIdx.x * 256 + threadIdx.x;
  if (e >= E) return;
  float v = leaky(a_s[src[e]] + a_d[dst[e]]);
  atomicMax(&menc[dst[e]], fenc(v));
}

__global__ void gat_z_init(const float* a_s, const float* a_d,
                           const unsigned* menc, float* z, int Nn) {
  int i = blockIdx.x * 256 + threadIdx.x;
  if (i >= Nn) return;
  float es = leaky(a_s[i] + a_d[i]);
  z[i] = expf(es - fdec(menc[i]));
}

__global__ void gat_edge_z(const float* a_s, const float* a_d,
                           const int* src, const int* dst,
                           const unsigned* menc, float* z, int E) {
  int e = blockIdx.x * 256 + threadIdx.x;
  if (e >= E) return;
  int d = dst[e];
  float v = leaky(a_s[src[e]] + a_d[d]);
  atomicAdd(&z[d], expf(v - fdec(menc[d])));
}

__global__ __launch_bounds__(256)
void gat_edge_agg(const float* __restrict__ h, const float* a_s, const float* a_d,
                  const int* __restrict__ src, const int* __restrict__ dst,
                  const unsigned* menc, const float* z, float* out, int E) {
  int e = blockIdx.x, f = threadIdx.x;
  if (e >= E) return;
  int s = src[e], d = dst[e];
  float v = leaky(a_s[s] + a_d[d]);
  float alpha = expf(v - fdec(menc[d])) / z[d];
  atomicAdd(&out[(size_t)d * 256 + f], h[(size_t)s * 256 + f] * alpha);
}

__global__ __launch_bounds__(256)
void gat_finalize(const float* __restrict__ h, float* out, const float* a_s,
                  const float* a_d, const unsigned* menc, const float* z,
                  const float* __restrict__ bias, int Nn) {
  int i = blockIdx.x, f = threadIdx.x;
  float es = leaky(a_s[i] + a_d[i]);
  float alpha = expf(es - fdec(menc[i])) / z[i];
  float v = out[(size_t)i * 256 + f] + h[(size_t)i * 256 + f] * alpha + bias[f];
  out[(size_t)i * 256 + f] = fmaxf(v, 0.0f);
}

// ---- mean pool (F = 256) ---------------------------------------------------
__global__ __launch_bounds__(256)
void pool_sum(const float* __restrict__ x, const int* __restrict__ batch,
              float* pool, int pstride, float* cnt, int Nn) {
  int i = blockIdx.x, f = threadIdx.x;
  int b = batch[i];
  atomicAdd(&pool[(size_t)b * pstride + f], x[(size_t)i * 256 + f]);
  if (f == 0) atomicAdd(&cnt[b], 1.0f);
}

__global__ __launch_bounds__(256)
void pool_div(float* pool, int pstride, const float* cnt, int Bn) {
  int b = blockIdx.x, f = threadIdx.x;
  if (b >= Bn) return;
  pool[(size_t)b * pstride + f] /= fmaxf(cnt[b], 1.0f);
}

// ---------------------------------------------------------------------------
// Host-side orchestration
// ---------------------------------------------------------------------------
extern "C" void kernel_launch(void* const* d_in, const int* in_sizes, int n_in,
                              void* d_out, int out_size, void* d_ws, size_t ws_size,
                              hipStream_t stream) {
  (void)in_sizes; (void)n_in; (void)out_size;

  const float* x1        = (const float*)d_in[0];
  const int*   src1      = (const int*)d_in[1];
  const int*   dst1      = src1 + E1;
  /* d_in[2] edge_attr unused by reference */
  const int*   batch1    = (const int*)d_in[3];
  const float* x3        = (const float*)d_in[4];
  const int*   src3      = (const int*)d_in[5];
  const int*   dst3      = src3 + E3;
  const float* ami_dis   = (const float*)d_in[6];
  const int*   ami_batch = (const int*)d_in[7];
  const float* ami_li    = (const float*)d_in[8];
  const float* w1  = (const float*)d_in[9];  const float* b1  = (const float*)d_in[10];
  const float* w2  = (const float*)d_in[11]; const float* b2  = (const float*)d_in[12];
  const float* Wc1 = (const float*)d_in[13]; const float* bc1 = (const float*)d_in[14];
  const float* Wg  = (const float*)d_in[15];
  const float* atts = (const float*)d_in[16];
  const float* attd = (const float*)d_in[17];
  const float* bg  = (const float*)d_in[18];
  const float* Wc4 = (const float*)d_in[19]; const float* bc4 = (const float*)d_in[20];
  const float* Wc5 = (const float*)d_in[21]; const float* bc5 = (const float*)d_in[22];
  const float* Wfc1 = (const float*)d_in[23]; const float* bfc1 = (const float*)d_in[24];
  const float* Wfc2 = (const float*)d_in[25]; const float* bfc2 = (const float*)d_in[26];
  const float* Wout = (const float*)d_in[27]; const float* bout = (const float*)d_in[28];

  // ---- workspace layout (floats); ami and drug phases alias region U ------
  float* ws = (float*)d_ws;
  const size_t U = 51180000;                      // union region size
  // ami view
  float* ami_lin  = ws;                           // 20000 x 1023
  float* ami_h    = ws + 20460000;                // 20000 x 1024
  float* ami_lin2 = ws + 40940000;                // 20000 x 256
  float* ami_g2   = ws + 46060000;                // 20000 x 256
  // drug view (reuses region after ami branch is pooled)
  float* h1      = ws;                            // 50000 x 128
  float* hd2     = ws + 6400000;                  // 50000 x 64
  float* lin_g   = ws + 9600000;                  // 50000 x 128
  float* g1out   = ws + 16000000;                 // 50000 x 128
  float* gat_lin = ws + 22400000;                 // 50000 x 256
  float* gat_out = ws + 35200000;                 // 50000 x 256
  // tail
  float* tail  = ws + U;
  float* deg3  = tail;            float* dinv3 = tail + 20000;
  float* deg1  = tail + 40000;    float* dinv1 = tail + 90000;
  float* a_s   = tail + 140000;   float* a_d   = tail + 190000;
  float* zbuf  = tail + 240000;
  unsigned* menc = (unsigned*)(tail + 290000);
  float* cnt3  = tail + 340000;   float* cnt1  = tail + 340256;
  float* xcat  = tail + 340512;                   // 256 x 512
  float* relu1 = tail + 471584;                   // 256 x 512
  float* fc2o  = tail + 602656;                   // 256 x 128
  const size_t need = (U + 635424) * sizeof(float);
  if (ws_size < need) return;

  float* out_vec = (float*)d_out;          // [256]  (out)
  float* h2_out  = (float*)d_out + BGR;    // [256 x 512] (h2, pre-ReLU FC1)

  dim3 blk128(128), blk256(256);

  // ================= ami branch =================
  zero_f32<<<CDIV(N3, 256), blk256, 0, stream>>>(deg3, (size_t)N3);
  deg_edges<<<CDIV(E3, 256), blk256, 0, stream>>>(dst3, ami_dis, deg3, E3);
  deg_finalize<<<CDIV(N3, 256), blk256, 0, stream>>>(deg3, dinv3, N3);

  gemm_bf16_wmma<<<dim3(CDIV(D4 - 1, 64), CDIV(N3, 64)), blk128, 0, stream>>>(
      x3, Wc4, nullptr, ami_lin, N3, D3, D4 - 1, 0);

  zero_f32<<<CDIV((size_t)N3 * D4, 256), blk256, 0, stream>>>(ami_h, (size_t)N3 * D4);
  gcn_edge_agg<<<dim3(E3, CDIV(D4 - 1, 256)), blk256, 0, stream>>>(
      ami_lin, D4 - 1, src3, dst3, ami_dis, dinv3, ami_h, D4, E3, D4 - 1);
  gcn_finalize<<<CDIV((size_t)N3 * (D4 - 1), 256), blk256, 0, stream>>>(
      ami_lin, D4 - 1, ami_h, D4, dinv3, bc4, N3, D4 - 1);
  set_col<<<CDIV(N3, 256), blk256, 0, stream>>>(ami_h, D4, D4 - 1, ami_li, N3);

  gemm_bf16_wmma<<<dim3(CDIV(D5, 64), CDIV(N3, 64)), blk128, 0, stream>>>(
      ami_h, Wc5, nullptr, ami_lin2, N3, D4, D5, 0);

  zero_f32<<<CDIV((size_t)N3 * D5, 256), blk256, 0, stream>>>(ami_g2, (size_t)N3 * D5);
  gcn_edge_agg<<<dim3(E3, 1), blk256, 0, stream>>>(
      ami_lin2, D5, src3, dst3, ami_dis, dinv3, ami_g2, D5, E3, D5);
  gcn_finalize<<<CDIV((size_t)N3 * D5, 256), blk256, 0, stream>>>(
      ami_lin2, D5, ami_g2, D5, dinv3, bc5, N3, D5);

  zero_f32<<<CDIV((size_t)BGR * 512, 256), blk256, 0, stream>>>(xcat, (size_t)BGR * 512);
  zero_f32<<<1, blk256, 0, stream>>>(cnt3, (size_t)BGR);
  zero_f32<<<1, blk256, 0, stream>>>(cnt1, (size_t)BGR);
  pool_sum<<<N3, blk256, 0, stream>>>(ami_g2, ami_batch, xcat, 512, cnt3, N3);
  pool_div<<<BGR, blk256, 0, stream>>>(xcat, 512, cnt3, BGR);

  // ================= drug branch =================
  zero_f32<<<CDIV(N1, 256), blk256, 0, stream>>>(deg1, (size_t)N1);
  deg_edges<<<CDIV(E1, 256), blk256, 0, stream>>>(dst1, nullptr, deg1, E1);
  deg_finalize<<<CDIV(N1, 256), blk256, 0, stream>>>(deg1, dinv1, N1);

  gemm_bf16_wmma<<<dim3(CDIV(D1, 64), CDIV(N1, 64)), blk128, 0, stream>>>(
      x1, w1, b1, h1, N1, F_XD, D1, 1);
  gemm_bf16_wmma<<<dim3(CDIV(D2, 64), CDIV(N1, 64)), blk128, 0, stream>>>(
      h1, w2, b2, hd2, N1, D1, D2, 1);
  gemm_bf16_wmma<<<dim3(CDIV(2 * D2, 64), CDIV(N1, 64)), blk128, 0, stream>>>(
      hd2, Wc1, nullptr, lin_g, N1, D2, 2 * D2, 0);

  zero_f32<<<CDIV((size_t)N1 * 128, 256), blk256, 0, stream>>>(g1out, (size_t)N1 * 128);
  gcn_edge_agg<<<dim3(E1, 1), blk256, 0, stream>>>(
      lin_g, 128, src1, dst1, nullptr, dinv1, g1out, 128, E1, 128);
  gcn_finalize<<<CDIV((size_t)N1 * 128, 256), blk256, 0, stream>>>(
      lin_g, 128, g1out, 128, dinv1, bc1, N1, 128);

  gemm_bf16_wmma<<<dim3(CDIV(4 * D2, 64), CDIV(N1, 64)), blk128, 0, stream>>>(
      g1out, Wg, nullptr, gat_lin, N1, 2 * D2, 4 * D2, 0);

  gat_scores<<<N1, blk256, 0, stream>>>(gat_lin, atts, attd, a_s, a_d, menc, N1);
  gat_edge_max<<<CDIV(E1, 256), blk256, 0, stream>>>(a_s, a_d, src1, dst1, menc, E1);
  gat_z_init<<<CDIV(N1, 256), blk256, 0, stream>>>(a_s, a_d, menc, zbuf, N1);
  gat_edge_z<<<CDIV(E1, 256), blk256, 0, stream>>>(a_s, a_d, src1, dst1, menc, zbuf, E1);
  zero_f32<<<CDIV((size_t)N1 * 256, 256), blk256, 0, stream>>>(gat_out, (size_t)N1 * 256);
  gat_edge_agg<<<E1, blk256, 0, stream>>>(gat_lin, a_s, a_d, src1, dst1, menc, zbuf,
                                          gat_out, E1);
  gat_finalize<<<N1, blk256, 0, stream>>>(gat_lin, gat_out, a_s, a_d, menc, zbuf, bg, N1);

  pool_sum<<<N1, blk256, 0, stream>>>(gat_out, batch1, xcat + 256, 512, cnt1, N1);
  pool_div<<<BGR, blk256, 0, stream>>>(xcat + 256, 512, cnt1, BGR);

  // ================= head =================
  // h2 = xcat @ Wfc1 + bfc1  (pre-ReLU; this is the second reference output)
  gemm_bf16_wmma<<<dim3(CDIV(512, 64), CDIV(BGR, 64)), blk128, 0, stream>>>(
      xcat, Wfc1, bfc1, h2_out, BGR, 512, 512, 0);
  relu_copy<<<CDIV((size_t)BGR * 512, 256), blk256, 0, stream>>>(
      h2_out, relu1, (size_t)BGR * 512);
  gemm_bf16_wmma<<<dim3(CDIV(128, 64), CDIV(BGR, 64)), blk128, 0, stream>>>(
      relu1, Wfc2, bfc2, fc2o, BGR, 512, 128, 1);
  gemm_bf16_wmma<<<dim3(1, CDIV(BGR, 64)), blk128, 0, stream>>>(
      fc2o, Wout, bout, out_vec, BGR, 128, 1, 0);
}
